// Graph_Attn_8409545966232
// MI455X (gfx1250) — matrix operand order repaired and checked by reading, not compiled, not run on hardware
//
#include <hip/hip_runtime.h>
#include <math.h>

// Problem constants (match the reference)
#define B_       64
#define NSENT    511
#define M_       512          // NSENT + 1
#define D_       512
#define LMBD     0.9f
#define EPSN     1e-8f
#define NITER    192          // 0.9^192 ~ 1.6e-9 -> below fp32 noise

typedef __attribute__((ext_vector_type(16))) _Float16 v16h;
typedef __attribute__((ext_vector_type(8)))  _Float16 v8h;
typedef __attribute__((ext_vector_type(8)))  float    v8f;

// ---------------------------------------------------------------------------
// Kernel 1: row-normalize topic = concat(cur_sent, hid_sents) -> f16 matrix N
// one 256-thread block per (b, row); each thread handles 2 of the 512 dims.
// ---------------------------------------------------------------------------
__global__ void normalize_rows(const float* __restrict__ hid,
                               const float* __restrict__ cur,
                               _Float16* __restrict__ nmat) {
    const int row = blockIdx.x;            // b*512 + m
    const int b   = row >> 9;
    const int m   = row & (M_ - 1);
    const float* src = (m == 0)
        ? (cur + (size_t)b * D_)
        : (hid + ((size_t)b * NSENT + (m - 1)) * D_);

    const int tid = threadIdx.x;           // 0..255
    float v0 = src[tid];
    float v1 = src[tid + 256];

    __shared__ float red[256];
    red[tid] = v0 * v0 + v1 * v1;
    __syncthreads();
    for (int o = 128; o > 0; o >>= 1) {
        if (tid < o) red[tid] += red[tid + o];
        __syncthreads();
    }
    const float scale = 1.0f / (sqrtf(red[0]) + EPSN);

    _Float16* dst = nmat + (size_t)row * D_;
    dst[tid]       = (_Float16)(v0 * scale);
    dst[tid + 256] = (_Float16)(v1 * scale);
}

// ---------------------------------------------------------------------------
// Kernel 2: W = |N * N^T| per batch via v_wmma_f32_16x16x32_f16.
// 2x2 register blocking: each wave owns a 32x32 output block (four 16x16
// tiles), so each K-step loads 2 A-frags + 2 B-frags and issues 4 WMMAs —
// 2x the arithmetic intensity of one-tile-per-wave.
// Fragment layouts per CDNA5 ISA 7.12.2 (wave32).
// ---------------------------------------------------------------------------
__device__ __forceinline__ v16h load_afrag(const _Float16* __restrict__ row,
                                           int kb, int sel) {
    // A 16x32 f16: halves[0..7] = K(kb+8*sel ..+8), halves[8..15] = K(kb+16+8*sel ..+8)
    v8h a0 = *(const v8h*)(row + kb + sel * 8);
    v8h a1 = *(const v8h*)(row + kb + 16 + sel * 8);
    v16h f;
#pragma unroll
    for (int i = 0; i < 8; ++i) { f[i] = a0[i]; f[i + 8] = a1[i]; }
    return f;
}

__global__ void gram_wmma(const _Float16* __restrict__ nmat,
                          float* __restrict__ W) {
    const int lane  = threadIdx.x & 31;
    const int wave  = threadIdx.x >> 5;
    const int block = blockIdx.x * (blockDim.x >> 5) + wave; // 0 .. B*256-1
    const int b     = block >> 8;                            // 256 blocks/batch
    const int t     = block & 255;
    const int mt2   = t >> 4;                                // 32-row block
    const int nt2   = t & 15;                                // 32-col block
    const int lh    = lane & 15;
    const int sel   = lane >> 4;                             // half-wave select

    const _Float16* nb = nmat + (size_t)b * M_ * D_;
    const _Float16* ar0 = nb + (size_t)(mt2 * 32 +      lh) * D_; // A rows, tile 0
    const _Float16* ar1 = nb + (size_t)(mt2 * 32 + 16 + lh) * D_; // A rows, tile 1
    const _Float16* bc0 = nb + (size_t)(nt2 * 32 +      lh) * D_; // B cols (rows of N)
    const _Float16* bc1 = nb + (size_t)(nt2 * 32 + 16 + lh) * D_;

    v8f c00 = {}, c01 = {}, c10 = {}, c11 = {};
#pragma unroll 2
    for (int kk = 0; kk < 16; ++kk) {
        const int kb = kk * 32;
        v16h a0 = load_afrag(ar0, kb, sel);
        v16h a1 = load_afrag(ar1, kb, sel);
        // B 32x16 f16: same K order as A (halves[0..7] = K(kb+8*sel ..+8), halves[8..15] = K(kb+16+8*sel ..+8))
        v16h b0 = load_afrag(bc0, kb, sel);
        v16h b1 = load_afrag(bc1, kb, sel);

        c00 = __builtin_amdgcn_wmma_f32_16x16x32_f16(false, a0, false, b0, (short)0, c00, false, false);
        c01 = __builtin_amdgcn_wmma_f32_16x16x32_f16(false, a0, false, b1, (short)0, c01, false, false);
        c10 = __builtin_amdgcn_wmma_f32_16x16x32_f16(false, a1, false, b0, (short)0, c10, false, false);
        c11 = __builtin_amdgcn_wmma_f32_16x16x32_f16(false, a1, false, b1, (short)0, c11, false, false);
    }

    // C/D layout: VGPR r, lane l -> (M = r + 8*sel, N = l&15). Apply |.| on store.
    float* wb = W + (size_t)b * M_ * M_;
    {
        float* o00 = wb + (size_t)(mt2 * 32 + sel * 8) * M_ + (nt2 * 32 + lh);
#pragma unroll
        for (int r = 0; r < 8; ++r) {
            o00[(size_t)r * M_]      = __builtin_fabsf(c00[r]);   // tile (0,0)
            o00[(size_t)r * M_ + 16] = __builtin_fabsf(c01[r]);   // tile (0,1)
        }
        float* o10 = wb + (size_t)(mt2 * 32 + 16 + sel * 8) * M_ + (nt2 * 32 + lh);
#pragma unroll
        for (int r = 0; r < 8; ++r) {
            o10[(size_t)r * M_]      = __builtin_fabsf(c10[r]);   // tile (1,0)
            o10[(size_t)r * M_ + 16] = __builtin_fabsf(c11[r]);   // tile (1,1)
        }
    }
}

// ---------------------------------------------------------------------------
// Kernel 3: deg[b][j] = column sum of W (== row sum: W symmetric). Coalesced.
// ---------------------------------------------------------------------------
__global__ void colsum(const float* __restrict__ W, float* __restrict__ deg) {
    const int b = blockIdx.x, j = threadIdx.x;   // 512 threads
    const float* Wb = W + (size_t)b * M_ * M_;
    float s = 0.0f;
#pragma unroll 8
    for (int i = 0; i < M_; ++i) s += Wb[(size_t)i * M_ + j];
    deg[b * M_ + j] = s;
}

// ---------------------------------------------------------------------------
// Kernel 4: Neumann-series solve of (I - lmbd * W * D^-1) x = e0.
//   x <- e0 + lmbd * W D^-1 x.  Using W symmetry:
//   x_new[i] = e0_i + sum_j W[j][i] * (lmbd * x_j / deg_j)  -> coalesced cols.
// One 512-thread block per batch; W stays L2-resident (64 MB < 192 MB L2).
// f_curr = (1 - lmbd) * x.
// ---------------------------------------------------------------------------
__global__ void power_iter(const float* __restrict__ W,
                           const float* __restrict__ deg,
                           float* __restrict__ fcurr) {
    const int b = blockIdx.x;
    const int i = threadIdx.x;                   // column index, 0..511
    __shared__ float ys[M_];

    const float scl = LMBD / deg[b * M_ + i];
    const float e0  = (i == 0) ? 1.0f : 0.0f;
    const float* Wb = W + (size_t)b * M_ * M_;

    float x = e0;
    for (int t = 0; t < NITER; ++t) {
        ys[i] = x * scl;
        __syncthreads();
        float acc = e0;
#pragma unroll 8
        for (int j = 0; j < M_; ++j)
            acc += Wb[(size_t)j * M_ + i] * ys[j];
        __syncthreads();
        x = acc;
    }
    fcurr[b * M_ + i] = (1.0f - LMBD) * x;
}

// ---------------------------------------------------------------------------
// Kernel 5: attention + weighted sum.
//   scores = relu(f_curr - f_prev); attn = scores / sum(scores)
//   weighted[b][d] = sum_{i=0..510} hid[b][i][d] * attn[b][i+1]
// ---------------------------------------------------------------------------
__global__ void attn_out(const float* __restrict__ hid,
                         const float* __restrict__ fprev,
                         const float* __restrict__ fcurr,
                         float* __restrict__ weighted) {
    const int b = blockIdx.x, i = threadIdx.x;   // 512 threads
    __shared__ float sc[M_];
    __shared__ float red[M_];

    float s = fcurr[b * M_ + i] - fprev[b * M_ + i];
    s = s > 0.0f ? s : 0.0f;
    red[i] = s;
    __syncthreads();
    for (int o = 256; o > 0; o >>= 1) {
        if (i < o) red[i] += red[i + o];
        __syncthreads();
    }
    sc[i] = s / red[0];                          // attn (matches ref: no eps)
    __syncthreads();

    // thread i == output dim d; coalesced sweep over the 511 sentences
    const float* hb = hid + (size_t)b * NSENT * D_;
    float acc = 0.0f;
#pragma unroll 4
    for (int k = 0; k < NSENT; ++k)
        acc += hb[(size_t)k * D_ + i] * sc[k + 1];
    weighted[b * D_ + i] = acc;
}

// ---------------------------------------------------------------------------
// Launch: outputs = [weighted (B*512) | f_curr (B*512)] flat in d_out.
// Workspace: [N f16: 32MB][W f32: 64MB][deg: 128KB]
// ---------------------------------------------------------------------------
extern "C" void kernel_launch(void* const* d_in, const int* in_sizes, int n_in,
                              void* d_out, int out_size, void* d_ws, size_t ws_size,
                              hipStream_t stream) {
    const float* hid   = (const float*)d_in[0];  // [64,511,512]
    const float* cur   = (const float*)d_in[1];  // [64,1,512]
    const float* fprev = (const float*)d_in[2];  // [64,512]

    float* out      = (float*)d_out;
    float* weighted = out;                       // first B*512 elements
    float* fcurr    = out + (size_t)B_ * M_;     // second B*512 elements

    char* ws = (char*)d_ws;
    _Float16* nmat = (_Float16*)ws;                                // 32 MB
    float*    W    = (float*)(ws + (size_t)32 * 1024 * 1024);      // 64 MB
    float*    deg  = (float*)(ws + (size_t)96 * 1024 * 1024);      // 128 KB

    normalize_rows<<<B_ * M_, 256, 0, stream>>>(hid, cur, nmat);
    gram_wmma<<<(B_ * 256) / 8, 256, 0, stream>>>(nmat, W);        // 8 waves/blk
    colsum<<<B_, M_, 0, stream>>>(W, deg);
    power_iter<<<B_, M_, 0, stream>>>(W, deg, fcurr);
    attn_out<<<B_, M_, 0, stream>>>(hid, fprev, fcurr, weighted);
}
